// TtViTSelfAttention_33724083208330
// MI455X (gfx1250) — matrix operand order repaired and checked by reading, not compiled, not run on hardware
//
#include <hip/hip_runtime.h>
#include <hip/hip_bf16.h>
#include <stdint.h>

#define NUM_HEADS 12
#define HEAD_DIM  64
#define SEQ       1024
#define HID       768
#define BATCH     8

typedef __attribute__((ext_vector_type(16))) __bf16 bf16x16;
typedef __attribute__((ext_vector_type(8)))  float  f32x8;

// pack two fp32 -> packed bf16x2 in a uint (compiler selects v_cvt_pk_bf16_f32)
static __device__ __forceinline__ unsigned cvt2(float a, float b) {
  union { __bf16 h[2]; unsigned u; } un;
  un.h[0] = (__bf16)a; un.h[1] = (__bf16)b;
  return un.u;
}
static __device__ __forceinline__ unsigned short cvt1(float a) {
  union { __bf16 h; unsigned short u; } un;
  un.h = (__bf16)a; return un.u;
}

// assemble a 16-element bf16 WMMA fragment from two 16B chunks
static __device__ __forceinline__ bf16x16 mk_frag(uint4 a, uint4 b) {
  union { bf16x16 v; uint4 q[2]; } u;
  u.q[0] = a; u.q[1] = b;
  return u.v;
}

// gfx1250 async global->LDS copy (16B per lane), tracked by ASYNCcnt.
// Inline asm keeps it portable across ROCm-7.2 / amdgpu-toolchain builtin skews.
static __device__ __forceinline__ void async_b128(const unsigned short* g,
                                                  unsigned short* l) {
  unsigned laddr = (unsigned)(uintptr_t)l;  // low 32 bits of flat-shared = LDS addr
  asm volatile("global_load_async_to_lds_b128 %0, %1, off"
               :: "v"(laddr), "v"(g)
               : "memory");
}
static __device__ __forceinline__ void wait_async0() {
  asm volatile("s_wait_asynccnt 0x0" ::: "memory");
}

// ---------------------------------------------------------------------------
// Kernel 1: fused QKV projection.  y[m,o] = sum_k X[m,k] * W[o,k]
// Outputs (bf16): Q (pre-scaled by 1/sqrt(hd)), K -> [B, nh, S, hd];
// V -> transposed [B, nh, hd, S].
// Block tile 128(M) x 64(N) (one head's 64 cols), 8 waves in 4x2,
// each wave 32x32 (2x2 WMMA).  Epilogue stages the result tile in LDS and
// does coalesced b128 global stores.
// ---------------------------------------------------------------------------
__global__ __launch_bounds__(256) void qkv_gemm_kernel(
    const float* __restrict__ X,
    const float* __restrict__ Wq, const float* __restrict__ Wk,
    const float* __restrict__ Wv,
    unsigned short* __restrict__ qb, unsigned short* __restrict__ kb,
    unsigned short* __restrict__ vtb)
{
  __shared__ __align__(16) unsigned short xs[128][40];   // 80B row stride
  __shared__ __align__(16) unsigned short wsm[64][40];
  __shared__ __align__(16) unsigned short res[128 * 72]; // result staging

  const int t      = threadIdx.x;
  const int lane   = t & 31;
  const int wid    = t >> 5;
  const int wave_m = wid & 3;    // 4 bands of 32 rows
  const int wave_n = wid >> 2;   // 2 bands of 32 cols
  const int m0     = blockIdx.x * 128;
  const int n0     = blockIdx.y * 64;
  const int w      = blockIdx.z;  // 0=Q 1=K 2=V
  const float* W   = (w == 0) ? Wq : (w == 1) ? Wk : Wv;

  const int lr = lane & 15;
  const int lh = lane >> 4;

  f32x8 acc[2][2];
#pragma unroll
  for (int i = 0; i < 2; ++i)
#pragma unroll
    for (int j = 0; j < 2; ++j)
      acc[i][j] = f32x8{0.f,0.f,0.f,0.f,0.f,0.f,0.f,0.f};

  for (int k0 = 0; k0 < HID; k0 += 32) {
    // stage X tile 128x32 fp32 -> bf16 (packed converts, b64 LDS stores)
#pragma unroll
    for (int i = 0; i < 4; ++i) {
      int idx = t + i * 256;
      int row = idx >> 3;
      int c4  = (idx & 7) * 4;
      float4 xv = *(const float4*)&X[(size_t)(m0 + row) * HID + k0 + c4];
      uint2 pk; pk.x = cvt2(xv.x, xv.y); pk.y = cvt2(xv.z, xv.w);
      *(uint2*)&xs[row][c4] = pk;
    }
    // stage W tile 64x32
#pragma unroll
    for (int i = 0; i < 2; ++i) {
      int idx = t + i * 256;
      int row = idx >> 3;
      int c4  = (idx & 7) * 4;
      float4 wv = *(const float4*)&W[(size_t)(n0 + row) * HID + k0 + c4];
      uint2 pk; pk.x = cvt2(wv.x, wv.y); pk.y = cvt2(wv.z, wv.w);
      *(uint2*)&wsm[row][c4] = pk;
    }
    __syncthreads();

    // A fragments: lane half 0 -> K {0..7,16..23}, half 1 -> K {8..15,24..31}
    const int ka = lh * 8;
    bf16x16 afrag[2];
#pragma unroll
    for (int fm = 0; fm < 2; ++fm) {
      int row = wave_m * 32 + fm * 16 + lr;
      afrag[fm] = mk_frag(*(const uint4*)&xs[row][ka],
                          *(const uint4*)&xs[row][ka + 16]);
    }
    // B fragments: lane half 0 -> K {0..7,16..23}, half 1 -> K {8..15,24..31}
    const int kbo = lh * 8;
    bf16x16 bfrag[2];
#pragma unroll
    for (int fn = 0; fn < 2; ++fn) {
      int col = wave_n * 32 + fn * 16 + lr;
      bfrag[fn] = mk_frag(*(const uint4*)&wsm[col][kbo],
                          *(const uint4*)&wsm[col][kbo + 16]);
    }
#pragma unroll
    for (int fm = 0; fm < 2; ++fm)
#pragma unroll
      for (int fn = 0; fn < 2; ++fn)
        acc[fm][fn] = __builtin_amdgcn_wmma_f32_16x16x32_bf16(
            false, afrag[fm], false, bfrag[fn], (short)0, acc[fm][fn],
            false, false);
    __syncthreads();
  }

  const int   b  = m0 >> 10;        // block never crosses a batch boundary
  const int   s0 = m0 & 1023;
  const int   h  = n0 >> 6;         // one head per 64-col block
  const float sc = (w == 0) ? 0.125f : 1.0f;  // fold 1/sqrt(64) into Q

  if (w == 2) {
    // V: stage transposed [64 d][128 s]; accumulator is M-contiguous per lane
    // -> pack 8 rows into one b128 LDS store.
#pragma unroll
    for (int fm = 0; fm < 2; ++fm) {
#pragma unroll
      for (int fn = 0; fn < 2; ++fn) {
        int dl = wave_n * 32 + fn * 16 + lr;
        int sl = wave_m * 32 + fm * 16 + lh * 8;
        uint4 pk;
        pk.x = cvt2(acc[fm][fn][0], acc[fm][fn][1]);
        pk.y = cvt2(acc[fm][fn][2], acc[fm][fn][3]);
        pk.z = cvt2(acc[fm][fn][4], acc[fm][fn][5]);
        pk.w = cvt2(acc[fm][fn][6], acc[fm][fn][7]);
        *(uint4*)&res[dl * 136 + sl] = pk;   // 272B row stride (16B aligned)
      }
    }
    __syncthreads();
    unsigned short* dst =
        vtb + ((size_t)(b * NUM_HEADS + h) * HEAD_DIM) * SEQ + s0;
#pragma unroll
    for (int i = 0; i < 4; ++i) {
      int idx = t + i * 256;              // 1024 uint4
      int row = idx >> 4;                 // d 0..63
      int c   = (idx & 15) * 8;           // s chunk
      *(uint4*)&dst[(size_t)row * SEQ + c] = *(const uint4*)&res[row * 136 + c];
    }
  } else {
    // Q/K: stage [128 s][64 d] (transpose through LDS), then coalesced stores
#pragma unroll
    for (int fm = 0; fm < 2; ++fm) {
#pragma unroll
      for (int fn = 0; fn < 2; ++fn) {
        int nl = wave_n * 32 + fn * 16 + lr;
#pragma unroll
        for (int r = 0; r < 8; ++r) {
          int ml = wave_m * 32 + fm * 16 + lh * 8 + r;
          res[ml * 72 + nl] = cvt1(acc[fm][fn][r] * sc);
        }
      }
    }
    __syncthreads();
    unsigned short* dst =
        ((w == 0) ? qb : kb) +
        ((size_t)(b * NUM_HEADS + h) * SEQ + s0) * HEAD_DIM;
#pragma unroll
    for (int i = 0; i < 4; ++i) {
      int idx = t + i * 256;              // 1024 uint4
      int row = idx >> 3;                 // s 0..127
      int c   = (idx & 7) * 8;            // d chunk
      *(uint4*)&dst[(size_t)row * HEAD_DIM + c] = *(const uint4*)&res[row * 72 + c];
    }
  }
}

// ---------------------------------------------------------------------------
// Kernel 2: flash attention.  Block = (b,h) x 128 q rows; 8 waves x 16 q rows.
// KV loop in tiles of 64 keys; K tile + V^T tile staged with
// global_load_async_to_lds_b128 (ASYNCcnt).  Online softmax with shfl_xor
// row reductions; P re-laid out through per-wave LDS.  Q pre-scaled.
// ---------------------------------------------------------------------------
__global__ __launch_bounds__(256) void attn_kernel(
    const unsigned short* __restrict__ qb,
    const unsigned short* __restrict__ kb,
    const unsigned short* __restrict__ vtb,
    float* __restrict__ out)
{
  __shared__ __align__(16) unsigned short kt[64][72];     // K tile  [key][d]
  __shared__ __align__(16) unsigned short vt[64][72];     // V^T tile [d][key]
  __shared__ __align__(16) unsigned short pb[8][16][72];  // per-wave P tile

  const int t    = threadIdx.x;
  const int lane = t & 31;
  const int wid  = t >> 5;
  const int lr   = lane & 15;
  const int lh   = lane >> 4;
  const int bh   = blockIdx.x;                 // b*12 + h
  const int q0   = blockIdx.y * 128 + wid * 16;
  const int b    = bh / NUM_HEADS;
  const int h    = bh % NUM_HEADS;

  const unsigned short* qhead = qb  + (size_t)bh * SEQ * HEAD_DIM;
  const unsigned short* khead = kb  + (size_t)bh * SEQ * HEAD_DIM;
  const unsigned short* vhead = vtb + (size_t)bh * HEAD_DIM * SEQ;

  // Q A-fragments for the wave's 16 rows, two 32-wide d chunks (hd = 64)
  bf16x16 qfrag[2];
  {
    const unsigned short* qrow = qhead + (size_t)(q0 + lr) * HEAD_DIM;
    int ka = lh * 8;
    qfrag[0] = mk_frag(*(const uint4*)&qrow[ka],      *(const uint4*)&qrow[ka + 16]);
    qfrag[1] = mk_frag(*(const uint4*)&qrow[32 + ka], *(const uint4*)&qrow[32 + ka + 16]);
  }

  float m_run[8], l_run[8];
  f32x8 oacc[4];
#pragma unroll
  for (int r = 0; r < 8; ++r) { m_run[r] = -1e30f; l_run[r] = 0.f; }
#pragma unroll
  for (int dt = 0; dt < 4; ++dt)
    oacc[dt] = f32x8{0.f,0.f,0.f,0.f,0.f,0.f,0.f,0.f};

  for (int kv0 = 0; kv0 < SEQ; kv0 += 64) {
    // async staging: 64x64 bf16 each = 512 x 16B; 2 per thread per tile
#pragma unroll
    for (int i = 0; i < 2; ++i) {
      int idx = t + i * 256;
      int row = idx >> 3;
      int c   = (idx & 7) * 8;
      async_b128(&khead[(size_t)(kv0 + row) * HEAD_DIM + c], &kt[row][c]);
      async_b128(&vhead[(size_t)row * SEQ + kv0 + c],        &vt[row][c]);
    }
    wait_async0();
    __syncthreads();

    // scores: 4 key-tiles of 16, contract over 2 d-chunks of 32
    f32x8 sacc[4];
#pragma unroll
    for (int nt = 0; nt < 4; ++nt)
      sacc[nt] = f32x8{0.f,0.f,0.f,0.f,0.f,0.f,0.f,0.f};
#pragma unroll
    for (int nt = 0; nt < 4; ++nt) {
#pragma unroll
      for (int dc = 0; dc < 2; ++dc) {
        int key   = nt * 16 + lr;
        int dbase = dc * 32 + lh * 8;
        bf16x16 bf = mk_frag(*(const uint4*)&kt[key][dbase],
                             *(const uint4*)&kt[key][dbase + 16]);
        sacc[nt] = __builtin_amdgcn_wmma_f32_16x16x32_bf16(
            false, qfrag[dc], false, bf, (short)0, sacc[nt], false, false);
      }
    }

    // online softmax (row = lh*8 + r; reduction over 16 lanes of each half)
#pragma unroll
    for (int r = 0; r < 8; ++r) {
      float mx = fmaxf(fmaxf(sacc[0][r], sacc[1][r]),
                       fmaxf(sacc[2][r], sacc[3][r]));
#pragma unroll
      for (int off = 1; off < 16; off <<= 1)
        mx = fmaxf(mx, __shfl_xor(mx, off, 32));
      float mnew  = fmaxf(m_run[r], mx);
      float alpha = __expf(m_run[r] - mnew);
      m_run[r] = mnew;
      float rs = 0.f;
#pragma unroll
      for (int nt = 0; nt < 4; ++nt) {
        float p = __expf(sacc[nt][r] - mnew);
        sacc[nt][r] = p;
        rs += p;
      }
#pragma unroll
      for (int off = 1; off < 16; off <<= 1)
        rs += __shfl_xor(rs, off, 32);
      l_run[r] = l_run[r] * alpha + rs;
#pragma unroll
      for (int dt = 0; dt < 4; ++dt) oacc[dt][r] *= alpha;
    }

    // P (C-layout) -> per-wave LDS -> A-layout (same-wave DS ops in-order)
#pragma unroll
    for (int nt = 0; nt < 4; ++nt)
#pragma unroll
      for (int r = 0; r < 8; ++r)
        pb[wid][lh * 8 + r][nt * 16 + lr] = cvt1(sacc[nt][r]);

    // O += P * V   (contract over 64 keys in 2 chunks of 32)
#pragma unroll
    for (int kc = 0; kc < 2; ++kc) {
      int ka = kc * 32 + lh * 8;
      bf16x16 pA = mk_frag(*(const uint4*)&pb[wid][lr][ka],
                           *(const uint4*)&pb[wid][lr][ka + 16]);
#pragma unroll
      for (int dt = 0; dt < 4; ++dt) {
        int d     = dt * 16 + lr;
        int kbase = kc * 32 + lh * 8;
        bf16x16 vB = mk_frag(*(const uint4*)&vt[d][kbase],
                             *(const uint4*)&vt[d][kbase + 16]);
        oacc[dt] = __builtin_amdgcn_wmma_f32_16x16x32_bf16(
            false, pA, false, vB, (short)0, oacc[dt], false, false);
      }
    }
    __syncthreads();
  }

  // normalize and store fp32 ctx [B, S, H] (lanes 0..15 cover 16 contiguous d)
#pragma unroll
  for (int dt = 0; dt < 4; ++dt) {
    int d = dt * 16 + lr;
#pragma unroll
    for (int r = 0; r < 8; ++r) {
      int s = q0 + lh * 8 + r;
      out[((size_t)(b * SEQ + s)) * HID + h * HEAD_DIM + d] =
          oacc[dt][r] / l_run[r];
    }
  }
}

// ---------------------------------------------------------------------------
extern "C" void kernel_launch(void* const* d_in, const int* in_sizes, int n_in,
                              void* d_out, int out_size, void* d_ws,
                              size_t ws_size, hipStream_t stream) {
  (void)in_sizes; (void)n_in; (void)out_size; (void)ws_size;
  const float* X  = (const float*)d_in[0];
  const float* Wq = (const float*)d_in[1];
  const float* Wk = (const float*)d_in[2];
  const float* Wv = (const float*)d_in[3];
  float* out = (float*)d_out;

  const size_t per = (size_t)BATCH * NUM_HEADS * SEQ * HEAD_DIM;  // elements
  unsigned short* qb  = (unsigned short*)d_ws;
  unsigned short* kb  = qb + per;
  unsigned short* vtb = kb + per;

  dim3 g1(64, 12, 3), b1(256);  // M/128, H/64, {Q,K,V}
  qkv_gemm_kernel<<<g1, b1, 0, stream>>>(X, Wq, Wk, Wv, qb, kb, vtb);

  dim3 g2(BATCH * NUM_HEADS, SEQ / 128), b2(256);
  attn_kernel<<<g2, b2, 0, stream>>>(qb, kb, vtb, out);
}
